// SpatialConv_8873402433647
// MI455X (gfx1250) — matrix operand order repaired and checked by reading, not compiled, not run on hardware
//
#include <hip/hip_runtime.h>
#include <hip/hip_bf16.h>

typedef __attribute__((ext_vector_type(16))) __bf16 v16bf;
typedef __attribute__((ext_vector_type(8)))  __bf16 v8bf;
typedef __attribute__((ext_vector_type(4)))  __bf16 v4bf;
typedef __attribute__((ext_vector_type(2)))  __bf16 v2bf;
typedef __attribute__((ext_vector_type(8)))  float  v8f;
typedef __attribute__((ext_vector_type(2)))  float  v2f;

#define B_  8
#define H_  128
#define W_  256
#define C_  128
#define LDF 132   // f32 LDS row stride (4-float pad => bank-shift per row)
#define LDA 136   // bf16 LDS row stride (8-bf16 pad)

#define K1_LDS (W_*LDF*4 + 9*C_*4)                 // 139776 B
#define K3_LDS (W_*LDF*4 + W_*LDA*2 + W_*LDA*2)    // 274432 B (<= 320KB WGP LDS)

// ------------------------------------------------------------------
// K0: pack center tap w2[0,4,:,:] into wave32 WMMA B-operand layout
//     (n = lane%16, k = kb*32 + (e/8)*16 + (lane/16)*8 + e%8), plus S1[j]=sum_c w1[j,c]
// ------------------------------------------------------------------
__global__ void k0_prepare(const float* __restrict__ w1,
                           const float* __restrict__ w2,
                           __bf16* __restrict__ Bpack,
                           float* __restrict__ S1) {
  int tid = threadIdx.x;
  if (tid < 9) {
    float s = 0.f;
    for (int c = 0; c < C_; ++c) s += w1[tid * C_ + c];
    S1[tid] = s;
  }
  for (int idx = tid; idx < C_ * C_; idx += blockDim.x) {
    int e    = idx & 15;
    int lane = (idx >> 4) & 31;
    int kb   = (idx >> 9) & 3;
    int nt   = idx >> 11;
    int k = kb * 32 + (e >> 3) * 16 + (lane >> 4) * 8 + (e & 7);
    int n = nt * 16 + (lane & 15);
    Bpack[idx] = (__bf16)w2[(4 * C_ + k) * C_ + n];   // w2 HWIO [1,9,C,D], tap 4
  }
}

// ------------------------------------------------------------------
// K1: s = c1(x) (pre-relu, incl. b1), t = relu(s). One WG per (b,h) row.
// ------------------------------------------------------------------
__global__ void __launch_bounds__(256, 1)
k1_conv1(const float* __restrict__ x, const float* __restrict__ w1,
         const float* __restrict__ b1,
         float* __restrict__ s_out, float* __restrict__ t_out) {
  extern __shared__ char smem[];
  float* ldsX = (float*)smem;                  // W_*LDF
  float* ldsW = (float*)(smem + W_ * LDF * 4); // 9*C_
  int tid = threadIdx.x;
  int b = blockIdx.x / H_, h = blockIdx.x % H_;
  const float* xrow = x + (size_t)(b * H_ + h) * W_ * C_;

  for (int i4 = tid; i4 < W_ * C_ / 4; i4 += 256) {
    float4 v = ((const float4*)xrow)[i4];
    int idx = i4 * 4;
    float* d = &ldsX[(idx >> 7) * LDF + (idx & 127)];
    d[0] = v.x; d[1] = v.y; d[2] = v.z; d[3] = v.w;
  }
  for (int i = tid; i < 9 * C_; i += 256) ldsW[i] = w1[i];
  __syncthreads();

  int w = tid;                                  // one pixel per thread
  float acc = b1[0];
  #pragma unroll
  for (int j = 0; j < 9; ++j) {
    int wi = w + j - 4;
    if (wi >= 0 && wi < W_) {
      const float* xr = &ldsX[wi * LDF];
      const float* wr = &ldsW[j * C_];
      float a0 = 0.f, a1 = 0.f, a2 = 0.f, a3 = 0.f;
      for (int c = 0; c < C_; c += 4) {
        a0 += xr[c]     * wr[c];
        a1 += xr[c + 1] * wr[c + 1];
        a2 += xr[c + 2] * wr[c + 2];
        a3 += xr[c + 3] * wr[c + 3];
      }
      acc += (a0 + a1) + (a2 + a3);
    }
  }
  size_t o = (size_t)(b * H_ + h) * W_ + w;
  s_out[o] = acc;
  t_out[o] = fmaxf(acc, 0.f);
}

// ------------------------------------------------------------------
// K2: scalar field a[b,i,w] such that z[b,i,w,:] = x[b,ip(i),w,:] + a[b,i,w]
//     u = relu(s + Sum_j t[prev_row, w+j-4]*S1[j])   (conv linearity)
// ------------------------------------------------------------------
__global__ void k2_field(const float* __restrict__ s_in,
                         const float* __restrict__ t_in,
                         const float* __restrict__ S1g,
                         float* __restrict__ a_out) {
  int b = blockIdx.x / H_, i = blockIdx.x % H_;
  int w = threadIdx.x;
  const float* sb = s_in + (size_t)b * H_ * W_;
  const float* tb = t_in + (size_t)b * H_ * W_;
  float av;
  if (i == H_ - 1) {
    av = tb[(H_ - 2) * W_ + w];                          // z[H-1] = y[H-1]
  } else {
    int ru = (i == H_ - 2) ? (H_ - 1) : i;               // off-by-one: row H-2 uses u[H-1]
    float uv = sb[ru * W_ + w];
    if (ru > 0) {
      const float* tr = &tb[(ru - 1) * W_];
      #pragma unroll
      for (int j = 0; j < 9; ++j) {
        int wi = w + j - 4;
        if (wi >= 0 && wi < W_) uv += tr[wi] * S1g[j];
      }
    }
    uv = fmaxf(uv, 0.f);
    float tp = 0.f;
    if (i == 0)            tp = tb[(H_ - 2) * W_ + w];   // wrap: y[H-1] term
    else if (i == H_ - 2)  tp = tb[(i - 1) * W_ + w];
    else if (i >= 2)       tp = tb[(i - 2) * W_ + w];
    av = uv + tp;
  }
  a_out[(size_t)(b * H_ + i) * W_ + w] = av;
}

// ------------------------------------------------------------------
// K3: per-row fused pass3+pass4.  256 thr = 8 wave32; wave nt owns one
// 16-wide N block (B frags loaded once, reused across all M-tiles and
// BOTH GEMMs since c2 weights are identical).  M-loop unrolled x2 so
// each wave keeps two independent WMMA dependency chains in flight
// (only 2 waves/SIMD at 268KB LDS -> ILP must cover XDL latency).
// ------------------------------------------------------------------
union AF { v16bf v; v8bf hh[2]; };

static __device__ inline v16bf load_afrag(const __bf16* pA) {
  AF af;
  af.hh[0] = *(const v8bf*)pA;            // K chunk [half*8 .. +8)
  af.hh[1] = *(const v8bf*)(pA + 16);     // K chunk [16+half*8 .. +8)
  return af.v;
}

__global__ void __launch_bounds__(256, 1)
k3_gemm(const float* __restrict__ x, const float* __restrict__ a_field,
        const __bf16* __restrict__ Bpack, const float* __restrict__ b2,
        float* __restrict__ out) {
  extern __shared__ char smem[];
  float*  ldsF = (float*)smem;                                    // z -> p (f32, exact)
  __bf16* ldsA = (__bf16*)(smem + W_ * LDF * 4);                  // z -> p (bf16 A operand)
  __bf16* ldsV = (__bf16*)(smem + W_ * LDF * 4 + W_ * LDA * 2);   // v = relu(c2(z))

  int tid = threadIdx.x;
  int b = blockIdx.x / H_, h = blockIdx.x % H_;
  int hp = (h == 0) ? (H_ - 1) : ((h <= H_ - 3) ? h - 1 : h);     // ip(h)
  const float* xrow = x + (size_t)(b * H_ + hp) * W_ * C_;
  const float* arow = a_field + (size_t)(b * H_ + h) * W_;

  // phase 0: z = x[ip(h)] + a  (f32 + packed bf16 copies)
  for (int i4 = tid; i4 < W_ * C_ / 4; i4 += 256) {
    int idx = i4 * 4;
    int w = idx >> 7, c = idx & 127;
    float av = arow[w];
    float4 xv = ((const float4*)xrow)[i4];
    float z0 = xv.x + av, z1 = xv.y + av, z2 = xv.z + av, z3 = xv.w + av;
    float* f = &ldsF[w * LDF + c];
    f[0] = z0; f[1] = z1; f[2] = z2; f[3] = z3;
    v4bf g = { (__bf16)z0, (__bf16)z1, (__bf16)z2, (__bf16)z3 };
    *(v4bf*)&ldsA[w * LDA + c] = g;                                // one b64 store
  }
  __syncthreads();

  int wave = tid >> 5, lane = tid & 31;
  int half = lane >> 4, lan16 = lane & 15;
  int n = wave * 16 + lan16;
  float bias = b2[n];

  v16bf bfrag[4];
  #pragma unroll
  for (int kb = 0; kb < 4; ++kb)
    bfrag[kb] = *(const v16bf*)&Bpack[((wave * 4 + kb) * 32 + lane) * 16];

  // GEMM1: v = relu(z @ W2 + b2) -> ldsV (bf16); two tiles in flight
  for (int mt = 0; mt < 16; mt += 2) {
    v8f acc0 = {}, acc1 = {};
    #pragma unroll
    for (int kb = 0; kb < 4; ++kb) {
      const __bf16* pA = &ldsA[(mt * 16 + lan16) * LDA + kb * 32 + half * 8];
      v16bf a0 = load_afrag(pA);
      v16bf a1 = load_afrag(pA + 16 * LDA);
      acc0 = __builtin_amdgcn_wmma_f32_16x16x32_bf16(false, a0, false, bfrag[kb],
                                                     (short)0, acc0, false, false);
      acc1 = __builtin_amdgcn_wmma_f32_16x16x32_bf16(false, a1, false, bfrag[kb],
                                                     (short)0, acc1, false, false);
    }
    #pragma unroll
    for (int r = 0; r < 8; ++r) {
      int m = mt * 16 + half * 8 + r;
      ldsV[m * LDA + n]            = (__bf16)fmaxf(acc0[r] + bias, 0.f);
      ldsV[(m + 16) * LDA + n]     = (__bf16)fmaxf(acc1[r] + bias, 0.f);
    }
  }
  __syncthreads();

  // phase 2: p[w] = z[w] + v[w-1]  (in-place f32, refresh bf16 A operand)
  // vectorized by 2 channels: b64 f32 ops, b32 bf16 ops
  for (int i2 = tid; i2 < W_ * C_ / 2; i2 += 256) {
    int idx = i2 * 2;
    int w = idx >> 7, c = idx & 127;
    v2f pv = *(v2f*)&ldsF[w * LDF + c];
    if (w > 0) {
      v2bf vv = *(v2bf*)&ldsV[(w - 1) * LDA + c];
      pv.x += (float)vv.x;
      pv.y += (float)vv.y;
    }
    *(v2f*)&ldsF[w * LDF + c] = pv;
    v2bf pb = { (__bf16)pv.x, (__bf16)pv.y };
    *(v2bf*)&ldsA[w * LDA + c] = pb;
  }
  __syncthreads();

  // GEMM2 + pass-4 epilogue straight from accumulators.
  // r[W-1] (needed by q[W-2]) is acc[7] of the SAME lane that owns q[W-2] (acc[6]).
  float* orow = out + (size_t)(b * H_ + h) * W_ * C_;
  for (int mt = 0; mt < 16; mt += 2) {
    v8f acc0 = {}, acc1 = {};
    #pragma unroll
    for (int kb = 0; kb < 4; ++kb) {
      const __bf16* pA = &ldsA[(mt * 16 + lan16) * LDA + kb * 32 + half * 8];
      v16bf a0 = load_afrag(pA);
      v16bf a1 = load_afrag(pA + 16 * LDA);
      acc0 = __builtin_amdgcn_wmma_f32_16x16x32_bf16(false, a0, false, bfrag[kb],
                                                     (short)0, acc0, false, false);
      acc1 = __builtin_amdgcn_wmma_f32_16x16x32_bf16(false, a1, false, bfrag[kb],
                                                     (short)0, acc1, false, false);
    }
    float rr0[8], rr1[8];
    #pragma unroll
    for (int r = 0; r < 8; ++r) {
      rr0[r] = fmaxf(acc0[r] + bias, 0.f);
      rr1[r] = fmaxf(acc1[r] + bias, 0.f);
    }
    #pragma unroll
    for (int r = 0; r < 8; ++r) {
      // tile mt (never contains W-1/W-2 since those live in tile 15 = odd)
      int w0 = mt * 16 + half * 8 + r;
      orow[(size_t)w0 * C_ + n] =
          rr0[r] + ldsF[((w0 + W_ - 1) & (W_ - 1)) * LDF + n];
      // tile mt+1
      int w1 = w0 + 16;
      float qv;
      if (w1 == W_ - 1)      qv = ldsF[(W_ - 1) * LDF + n];          // q[W-1] = p[W-1]
      else if (w1 == W_ - 2) qv = rr1[7] + ldsF[(W_ - 2) * LDF + n]; // r[W-1] + p[W-2]
      else                   qv = rr1[r] + ldsF[((w1 + W_ - 1) & (W_ - 1)) * LDF + n];
      orow[(size_t)w1 * C_ + n] = qv;
    }
  }
}

// ------------------------------------------------------------------
extern "C" void kernel_launch(void* const* d_in, const int* in_sizes, int n_in,
                              void* d_out, int out_size, void* d_ws, size_t ws_size,
                              hipStream_t stream) {
  (void)in_sizes; (void)n_in; (void)out_size; (void)ws_size;
  const float* x  = (const float*)d_in[0];
  const float* w1 = (const float*)d_in[1];
  const float* b1 = (const float*)d_in[2];
  const float* w2 = (const float*)d_in[3];
  const float* b2 = (const float*)d_in[4];
  float* out = (float*)d_out;

  char* ws = (char*)d_ws;
  float*  s_buf = (float*)(ws + 0);           // B*H*W f32 = 1 MB
  float*  t_buf = (float*)(ws + (1u << 20));  // 1 MB
  float*  a_buf = (float*)(ws + (2u << 20));  // 1 MB
  __bf16* Bpack = (__bf16*)(ws + (3u << 20)); // 32 KB
  float*  S1    = (float*)(ws + (3u << 20) + (32u << 10));

  // raise dynamic-LDS caps every call (deterministic, capture-safe: not a stream op)
  hipFuncSetAttribute(reinterpret_cast<const void*>(k1_conv1),
                      hipFuncAttributeMaxDynamicSharedMemorySize, K1_LDS);
  hipFuncSetAttribute(reinterpret_cast<const void*>(k3_gemm),
                      hipFuncAttributeMaxDynamicSharedMemorySize, K3_LDS);

  k0_prepare<<<1, 256, 0, stream>>>(w1, w2, Bpack, S1);
  k1_conv1 <<<B_ * H_, 256, K1_LDS, stream>>>(x, w1, b1, s_buf, t_buf);
  k2_field <<<B_ * H_, W_, 0, stream>>>(s_buf, t_buf, S1, a_buf);
  k3_gemm  <<<B_ * H_, 256, K3_LDS, stream>>>(x, a_buf, Bpack, b2, out);
}
